// MHKPattLayer_73031623901709
// MI455X (gfx1250) — matrix operand order repaired and checked by reading, not compiled, not run on hardware
//
#include <hip/hip_runtime.h>
#include <math.h>

// ---- problem constants (from reference) ----
#define BSZ   4
#define TSEQ  2048
#define CDIM  1024
#define HEADS 8
#define NTOK  1024
#define DK    128
#define DV    128

#define WAVES       4
#define BLOCK_THR   128
#define ROWS_PER_WV 16
#define BLOCK_ROWS  (WAVES * ROWS_PER_WV)   // 64

typedef __attribute__((ext_vector_type(16))) __bf16 v16bf;
typedef __attribute__((ext_vector_type(2)))  __bf16 bf16x2;
typedef __attribute__((ext_vector_type(8)))  float  v8f;
typedef __attribute__((ext_vector_type(4)))  int    v4i;

union BF16x16 {
    v16bf          v;
    unsigned short u[16];
    unsigned int   d[8];
};

// ---- native bf16 converts (v_cvt_pk_bf16_f32 path) ----
__device__ __forceinline__ unsigned int pack_bf16(float a, float b) {
    union { bf16x2 v; unsigned int u; } cv;
    cv.v[0] = (__bf16)a;
    cv.v[1] = (__bf16)b;
    return cv.u;
}
__device__ __forceinline__ unsigned short f2bf(float f) {
    union { __bf16 v; unsigned short u; } cv;
    cv.v = (__bf16)f;
    return cv.u;
}
__device__ __forceinline__ float bf2f(unsigned short h) {
    union { unsigned short u; __bf16 v; } cv;
    cv.u = h;
    return (float)cv.v;
}
__device__ __forceinline__ float gelu_exact(float x) {
    return 0.5f * x * (1.0f + erff(x * 0.70710678118654752f));
}

// ---- gfx1250 async global->LDS copy (ASYNCcnt path), with safe fallback ----
#if defined(__gfx1250__) && __has_builtin(__builtin_amdgcn_global_load_async_to_lds_b128)
#define HAVE_ASYNC 1
#else
#define HAVE_ASYNC 0
#endif

#if HAVE_ASYNC
typedef __attribute__((address_space(1))) v4i g_v4i;   // global int4*
typedef __attribute__((address_space(3))) v4i l_v4i;   // LDS int4*
__device__ __forceinline__ void async_copy16(const float* g, float* l) {
    __builtin_amdgcn_global_load_async_to_lds_b128(
        (g_v4i*)(void*)g, (l_v4i*)(void*)l, 0, 0);
}
__device__ __forceinline__ void wait_async0() {
#if __has_builtin(__builtin_amdgcn_s_wait_asynccnt)
    __builtin_amdgcn_s_wait_asynccnt(0);
#else
    asm volatile("s_wait_asynccnt 0x0" ::: "memory");
#endif
}
#endif

// contiguous fp32 stage into raw LDS buffer (async when available)
__device__ __forceinline__ void issue_stage(const float* __restrict__ src,
                                            float* rawbuf, int nfloats, int tid) {
    for (int i = tid * 4; i < nfloats; i += BLOCK_THR * 4) {
#if HAVE_ASYNC
        async_copy16(src + i, rawbuf + i);
#else
        *(float4*)&rawbuf[i] = *(const float4*)&src[i];
#endif
    }
}

__global__ __launch_bounds__(BLOCK_THR, 1)
void tokenformer_fused(const float* __restrict__ x,
                       const float* __restrict__ key_tokens,
                       const float* __restrict__ val_tokens,
                       float* __restrict__ out)
{
    __shared__ __align__(16) unsigned short qlds[BLOCK_ROWS][DK];          // 16 KB bf16
    __shared__ __align__(16) unsigned short klds[16][DK];                  //  4 KB bf16
    __shared__ __align__(16) unsigned short vlds[32][DV];                  //  8 KB bf16
    __shared__ __align__(16) unsigned short sco[WAVES][ROWS_PER_WV][NTOK]; // 128 KB scores
    __shared__ __align__(16) float raw[BLOCK_ROWS * DK];                   // 32 KB fp32 stage
    __shared__ float rowscale[WAVES][ROWS_PER_WV];
    __shared__ float rowmean[WAVES][ROWS_PER_WV];

    const int tid  = threadIdx.x;
    const int wid  = tid >> 5;
    const int lane = tid & 31;
    const int half = lane >> 4;
    const int l15  = lane & 15;

    const int bid  = blockIdx.x;          // 1024 = B*H*(T/64)
    const int bh   = bid >> 5;
    const int tile = bid & 31;
    const int b    = bh >> 3;
    const int h    = bh & 7;
    const int t0   = tile * BLOCK_ROWS;

    // ---------------- stage Q (async fp32 -> raw, packed convert -> bf16) ----------------
    const float* xq = x + ((size_t)b * TSEQ + t0) * CDIM + (size_t)h * DK;
    for (int i = tid * 4; i < BLOCK_ROWS * DK; i += BLOCK_THR * 4) {
        const float* src = xq + (size_t)(i >> 7) * CDIM + (i & 127);
#if HAVE_ASYNC
        async_copy16(src, &raw[i]);
#else
        *(float4*)&raw[i] = *(const float4*)src;
#endif
    }
#if HAVE_ASYNC
    wait_async0();
#endif
    __syncthreads();
    {
        unsigned short* qflat = &qlds[0][0];
        for (int i = tid * 4; i < BLOCK_ROWS * DK; i += BLOCK_THR * 4) {
            float4 f = *(const float4*)&raw[i];
            uint2 pk;
            pk.x = pack_bf16(f.x, f.y);
            pk.y = pack_bf16(f.z, f.w);
            *(uint2*)(qflat + i) = pk;
        }
    }
    __syncthreads();

    // ---------------- GEMM1 prologue: issue K tile 0 while building A-fragments ----------------
    const float* khead = key_tokens + (size_t)h * NTOK * DK;
    issue_stage(khead, raw, 16 * DK, tid);

    // A-fragments (16-bit A layout: lanes 0-15 -> K {0..7,16..23}; lanes 16-31 -> K {8..15,24..31})
    BF16x16 qf[4];
    {
        const int arow = wid * ROWS_PER_WV + l15;
        for (int kc = 0; kc < 4; ++kc) {
            const int kb = kc * 32 + half * 8;
            for (int p = 0; p < 4; ++p) {
                qf[kc].d[p]     = *(const unsigned int*)&qlds[arow][kb + 2 * p];
                qf[kc].d[p + 4] = *(const unsigned int*)&qlds[arow][kb + 16 + 2 * p];
            }
        }
    }

    // ---------------- GEMM1: scores = q @ K^T (software-pipelined tiles) ----------------
    float ss[8];
    for (int r = 0; r < 8; ++r) ss[r] = 0.0f;

    for (int nt = 0; nt < NTOK / 16; ++nt) {
#if HAVE_ASYNC
        wait_async0();
#endif
        __syncthreads();
        // cooperative packed convert raw(fp32) -> klds(bf16)
        {
            unsigned short* kflat = &klds[0][0];
            for (int i = tid * 4; i < 16 * DK; i += BLOCK_THR * 4) {
                float4 f = *(const float4*)&raw[i];
                uint2 pk;
                pk.x = pack_bf16(f.x, f.y);
                pk.y = pack_bf16(f.z, f.w);
                *(uint2*)(kflat + i) = pk;
            }
        }
        __syncthreads();
        // overlap: fetch next K tile while this one is consumed by WMMA
        if (nt + 1 < NTOK / 16)
            issue_stage(khead + (size_t)(nt + 1) * 16 * DK, raw, 16 * DK, tid);

        v8f acc;
        for (int i = 0; i < 8; ++i) acc[i] = 0.0f;
        for (int kc = 0; kc < 4; ++kc) {
            // B layout 32x16: VGPR r<4 -> K=8h+2r,+1; r>=4 -> K=16+8h+2(r-4),+1 (h = lane/16); col = lane%16
            BF16x16 bf;
            const int kb = kc * 32 + half * 8;
            for (int r = 0; r < 8; ++r)
                bf.d[r] = *(const unsigned int*)&klds[l15][kb + 2 * r + (r >= 4 ? 8 : 0)];
            acc = __builtin_amdgcn_wmma_f32_16x16x32_bf16(
                false, qf[kc].v, false, bf.v, (short)0, acc, false, false);
        }

        const int n = nt * 16 + l15;
        for (int r = 0; r < 8; ++r) {
            float s = acc[r];
            sco[wid][half ? (r + 8) : r][n] = f2bf(s);
            ss[r] += s * s;
        }
        __syncthreads();
    }

    // per-row sum-of-squares: reduce across the 16 lanes of each half-wave
    for (int m = 1; m < 16; m <<= 1)
        for (int r = 0; r < 8; ++r)
            ss[r] += __shfl_xor(ss[r], m, 32);
    if (l15 == 0) {
        for (int r = 0; r < 8; ++r) {
            float nrm = sqrtf(ss[r]);
            rowscale[wid][half ? (r + 8) : r] = 32.0f / fmaxf(nrm, 1e-8f); // sqrt(N)=32
        }
    }
    __syncthreads();

    // ---------------- GEMM2 prologue: stream V tile 0 in while doing GELU/mean pass ----------------
    const float* vhead = val_tokens + (size_t)h * NTOK * DV;
    issue_stage(vhead, raw, 32 * DV, tid);

    // theta: scale + exact GELU in place, then row mean
    for (int m = 0; m < ROWS_PER_WV; ++m) {
        const float scale = rowscale[wid][m];
        float accm = 0.0f;
        for (int n0 = lane * 2; n0 < NTOK; n0 += 64) {
            unsigned int pr = *(unsigned int*)&sco[wid][m][n0];
            float g0 = gelu_exact(bf2f((unsigned short)(pr & 0xFFFF)) * scale);
            float g1 = gelu_exact(bf2f((unsigned short)(pr >> 16)) * scale);
            *(unsigned int*)&sco[wid][m][n0] = pack_bf16(g0, g1);
            accm += g0 + g1;
        }
        for (int msk = 1; msk < 32; msk <<= 1) accm += __shfl_xor(accm, msk, 32);
        if (lane == 0) rowmean[wid][m] = accm * (1.0f / NTOK);
    }
    __syncthreads();

    // ---------------- GEMM2: out = hotmask(gelu_scores) @ V (pipelined tiles) ----------------
    v8f oacc[8];
    for (int dv = 0; dv < 8; ++dv)
        for (int i = 0; i < 8; ++i) oacc[dv][i] = 0.0f;

    const float meanrow = rowmean[wid][l15];

    for (int c = 0; c < NTOK / 32; ++c) {
#if HAVE_ASYNC
        wait_async0();
#endif
        __syncthreads();
        // cooperative packed convert raw(fp32) -> vlds(bf16)
        {
            unsigned short* vflat = &vlds[0][0];
            for (int i = tid * 4; i < 32 * DV; i += BLOCK_THR * 4) {
                float4 f = *(const float4*)&raw[i];
                uint2 pk;
                pk.x = pack_bf16(f.x, f.y);
                pk.y = pack_bf16(f.z, f.w);
                *(uint2*)(vflat + i) = pk;
            }
        }
        __syncthreads();
        if (c + 1 < NTOK / 32)
            issue_stage(vhead + (size_t)(c + 1) * 32 * DV, raw, 32 * DV, tid);

        // A fragment from gelu'd scores, hot-mask (> row mean) before packing
        BF16x16 af;
        const int nb = c * 32 + half * 8;
        for (int p = 0; p < 4; ++p) {
            af.d[p]     = *(const unsigned int*)&sco[wid][l15][nb + 2 * p];
            af.d[p + 4] = *(const unsigned int*)&sco[wid][l15][nb + 16 + 2 * p];
        }
        for (int e = 0; e < 16; ++e) {
            if (!(bf2f(af.u[e]) > meanrow)) af.u[e] = 0;
        }

        for (int dv = 0; dv < 8; ++dv) {
            BF16x16 bfv;
            const int kb = half * 8;
            for (int r = 0; r < 8; ++r) {
                unsigned int a0 = vlds[kb + 2 * r + (r >= 4 ? 8 : 0)][dv * 16 + l15];
                unsigned int a1 = vlds[kb + 2 * r + (r >= 4 ? 8 : 0) + 1][dv * 16 + l15];
                bfv.d[r] = a0 | (a1 << 16);
            }
            oacc[dv] = __builtin_amdgcn_wmma_f32_16x16x32_bf16(
                false, af.v, false, bfv.v, (short)0, oacc[dv], false, false);
        }
        __syncthreads();
    }

    // ---------------- write out [B, T, H*Dv] fp32 ----------------
    float* op = out + ((size_t)b * TSEQ + t0 + wid * ROWS_PER_WV) * (HEADS * DV) + (size_t)h * DV;
    for (int dv = 0; dv < 8; ++dv) {
        for (int r = 0; r < 8; ++r) {
            int row = half ? (r + 8) : r;
            op[(size_t)row * (HEADS * DV) + dv * 16 + l15] = oacc[dv][r];
        }
    }
}

extern "C" void kernel_launch(void* const* d_in, const int* in_sizes, int n_in,
                              void* d_out, int out_size, void* d_ws, size_t ws_size,
                              hipStream_t stream) {
    (void)in_sizes; (void)n_in; (void)out_size; (void)d_ws; (void)ws_size;
    const float* x  = (const float*)d_in[0];
    const float* kt = (const float*)d_in[1];
    const float* vt = (const float*)d_in[2];
    float* out = (float*)d_out;

    dim3 grid(BSZ * HEADS * (TSEQ / BLOCK_ROWS));  // 1024
    dim3 block(BLOCK_THR);                          // 128 = 4 waves (wave32)
    hipLaunchKernelGGL(tokenformer_fused, grid, block, 0, stream, x, kt, vt, out);
}
